// EarthAttention3D_19937238188537
// MI455X (gfx1250) — matrix operand order repaired and checked by reading, not compiled, not run on hardware
//
#include <hip/hip_runtime.h>

// ---------------------------------------------------------------------------
// EarthAttention3D for gfx1250 (MI455X): f16 WMMA pipeline.
//   K0: convert x / qkv_w / proj_w to f16 once (coalesced, exact tiling)
//   K1: qkv = xh @ wh^T + b (pure async-to-LDS staging, double-buffered,
//       fused q-scale; q/k row-major, V transposed)
//   K2: bias expand: biasbuf[w][h][q][k] = table[pos_index[q,k], w, h]  (f16)
//   K3: per-(b,head,window): QK^T + bias + mask + softmax + PV
//   K4: out = attn_out @ proj_wh^T + proj_b (same async GEMM, fp32 out)
// All WMMA operand loads are contiguous 16B/32B LDS reads matching the CDNA5
// 16-bit A/B VGPR striping (ds_load_b128). GEMM staging has no VALU at all:
// f16 tiles are DMA'd with global_load_async_to_lds_b128 into the back
// buffer while WMMAs consume the front buffer (ASYNCcnt + s_wait_asynccnt).
// ---------------------------------------------------------------------------

typedef __attribute__((ext_vector_type(16))) _Float16 v16h;
typedef __attribute__((ext_vector_type(8)))  _Float16 v8h;
typedef __attribute__((ext_vector_type(8)))  float    v8f;

#define AS1 __attribute__((address_space(1)))
#define AS3 __attribute__((address_space(3)))

// Exact parameter type of __builtin_amdgcn_global_load_async_to_lds_b128
// (per hipcc diagnostic): pointer to gcc-style int vector of 4.
typedef int ea_i4 __attribute__((vector_size(16)));

#define DIMC   192
#define HEADS  6
#define HD     32
#define NTOK   144
#define NW     64
#define BBATCH 15
#define NPAD   160                       // 144 padded to 5*32 for PV K-loop
#define TABLE_ROWS 3312
#define MROWS  (BBATCH * NW * NTOK)      // 138240
#define QKV_ELEMS  ((size_t)BBATCH * HEADS * NW * NTOK * HD)   // 26,542,080 h
#define BIAS_ELEMS ((size_t)NW * HEADS * NTOK * NTOK)          //  7,962,624 h

#define WMMA_F16(A, B, C) \
  __builtin_amdgcn_wmma_f32_16x16x32_f16(false, (A), false, (B), (short)0, (C), false, false)

#if defined(__has_builtin)
#if __has_builtin(__builtin_amdgcn_global_load_async_to_lds_b128)
#define EA3D_ASYNC 1
#endif
#endif
#ifndef EA3D_ASYNC
#define EA3D_ASYNC 0
#endif

#if EA3D_ASYNC
#if __has_builtin(__builtin_amdgcn_s_wait_asynccnt)
#define EA3D_WAIT_ASYNC() __builtin_amdgcn_s_wait_asynccnt(0)
#else
#define EA3D_WAIT_ASYNC() asm volatile("s_wait_asynccnt 0" ::: "memory")
#endif
#else
#define EA3D_WAIT_ASYNC() ((void)0)
#endif

// Global pointer -> typed AS1 pointer (same 64-bit VA; inttoptr).
static __device__ __forceinline__ AS1 ea_i4* gcast(const void* p) {
  return (AS1 ea_i4*)(unsigned long long)p;
}
// Flat LDS pointer -> typed AS3 pointer. ISA 10.2: LDS aperture access
// truncates the flat address to addr[31:0] -> low 32 bits are the DS address.
static __device__ __forceinline__ AS3 ea_i4* lcast(const void* p) {
  return (AS3 ea_i4*)(unsigned)(unsigned long long)p;
}

// A-matrix fragment (16x32 f16) from a row-major [16][lda] tile.
// Lane L: row = L&15, hi = L>>4. Halves 0-7 = K [8hi, 8hi+8),
// halves 8-15 = K [16+8hi, 16+8hi+8): two contiguous 16-byte LDS loads.
static __device__ __forceinline__ v16h frag_a(const _Float16* base, int lda, int lane) {
  const int row = lane & 15, hi = lane >> 4;
  const _Float16* p = base + row * lda + 8 * hi;
  union { v16h v; v8h h[2]; } u;
  u.h[0] = *(const v8h*)(p);
  u.h[1] = *(const v8h*)(p + 16);
  return u.v;
}

// B-matrix fragment (32x16 f16) from an n-major / k-contiguous tile:
// element (k, n) at base[n*stride_n + k]. Lane L: n = L&15, hi = L>>4.
// Halves 0-7 = K [8hi, 8hi+8), halves 8-15 = K [16+8hi, 16+8hi+8) (two ds_load_b128).
static __device__ __forceinline__ v16h frag_b(const _Float16* base, int stride_n, int lane) {
  const int n = lane & 31 & 15, hi = lane >> 4;
  const _Float16* p = base + (lane & 15) * stride_n + 8 * hi;
  union { v16h v; v8h h[2]; } u;
  u.h[0] = *(const v8h*)(p);
  u.h[1] = *(const v8h*)(p + 16);
  return u.v;
}

// ---------------------------------------------------------------------------
// Kernel 0: f32 -> f16 conversion, 8 elements/thread, exact tiling.
// ---------------------------------------------------------------------------
__global__ __launch_bounds__(256) void ea3d_cvt_kernel(
    const float* __restrict__ src, _Float16* __restrict__ dst) {
  const size_t i = ((size_t)blockIdx.x * 256 + threadIdx.x) * 8;
  const float4 f0 = *(const float4*)(src + i);
  const float4 f1 = *(const float4*)(src + i + 4);
  v8h h;
  h[0] = (_Float16)f0.x; h[1] = (_Float16)f0.y;
  h[2] = (_Float16)f0.z; h[3] = (_Float16)f0.w;
  h[4] = (_Float16)f1.x; h[5] = (_Float16)f1.y;
  h[6] = (_Float16)f1.z; h[7] = (_Float16)f1.w;
  *(v8h*)(dst + i) = h;
}

// ---------------------------------------------------------------------------
// Kernel 1: QKV projection (f16 in). Tiles: 64(M) x 64(N), 4 waves, K=192 in
// 6 steps. Both tiles DMA'd async into the back LDS buffer during compute.
// ---------------------------------------------------------------------------
__global__ __launch_bounds__(128) void ea3d_qkv_kernel(
    const _Float16* __restrict__ xh, const _Float16* __restrict__ wh,
    const float* __restrict__ bias,
    _Float16* __restrict__ qb, _Float16* __restrict__ kb, _Float16* __restrict__ vb) {
  __shared__ __align__(32) _Float16 As[2][64 * 32];   // row-major [m][k]
  __shared__ __align__(32) _Float16 Bs[2][64 * 32];   // n-major  [n][k]
  const int tid = threadIdx.x, lane = tid & 31, wv = tid >> 5;
  const int rowBase = blockIdx.x * 64;
  const int colBase = blockIdx.y * 64;

  auto stage = [&](int kt, int buf) {
#pragma unroll
    for (int it = 0; it < 2; ++it) {      // 256 chunks per tile, 2+2/thread
      const int i = tid + it * 128;
      const int r = i >> 2, c8 = (i & 3) * 8;
#if EA3D_ASYNC
      __builtin_amdgcn_global_load_async_to_lds_b128(
          gcast(xh + (size_t)(rowBase + r) * DIMC + kt * 32 + c8),
          lcast(&As[buf][r * 32 + c8]), 0, 0);
      __builtin_amdgcn_global_load_async_to_lds_b128(
          gcast(wh + (size_t)(colBase + r) * DIMC + kt * 32 + c8),
          lcast(&Bs[buf][r * 32 + c8]), 0, 0);
#else
      *(v8h*)&As[buf][r * 32 + c8] =
          *(const v8h*)(xh + (size_t)(rowBase + r) * DIMC + kt * 32 + c8);
      *(v8h*)&Bs[buf][r * 32 + c8] =
          *(const v8h*)(wh + (size_t)(colBase + r) * DIMC + kt * 32 + c8);
#endif
    }
  };

  v8f z = {};
  v8f acc[4];
#pragma unroll
  for (int nt = 0; nt < 4; ++nt) acc[nt] = z;

  stage(0, 0);
  EA3D_WAIT_ASYNC();
  __syncthreads();
#pragma unroll
  for (int kt = 0; kt < 6; ++kt) {
    const int buf = kt & 1;
    if (kt < 5) stage(kt + 1, buf ^ 1);   // DMA overlaps WMMA below
    v16h a = frag_a(&As[buf][wv * 16 * 32], 32, lane);
#pragma unroll
    for (int nt = 0; nt < 4; ++nt) {
      v16h b = frag_b(&Bs[buf][nt * 16 * 32], 32, lane);
      acc[nt] = WMMA_F16(a, b, acc[nt]);
    }
    if (kt < 5) { EA3D_WAIT_ASYNC(); __syncthreads(); }
  }

  // ---- epilogue: part/dst/scale are block-uniform (64 | 192) ----
  const int hi = lane >> 4, nn = lane & 15;
  const int part = colBase / DIMC;                 // 0=q 1=k 2=v
  _Float16* const dst = (part == 0) ? qb : (part == 1) ? kb : vb;
  const float scl = (part == 0) ? 0.17677669529663687f : 1.0f;  // hd^-0.5 on q
  float bv[4]; int headv[4], hdiv[4];
#pragma unroll
  for (int nt = 0; nt < 4; ++nt) {
    const int cg = colBase + nt * 16 + nn;
    const int c  = cg % DIMC;
    headv[nt] = c >> 5; hdiv[nt] = c & 31;
    bv[nt] = bias[cg];
  }
  // one decomposition + incremental carry (no per-i div/mod)
  const int rg0 = rowBase + wv * 16 + 8 * hi;
  int b_  = rg0 / (NW * NTOK);
  int rem = rg0 % (NW * NTOK);
  int ww = rem / NTOK, ntok = rem % NTOK;
#pragma unroll
  for (int i = 0; i < 8; ++i) {
#pragma unroll
    for (int nt = 0; nt < 4; ++nt) {
      const size_t base = (((size_t)b_ * HEADS + headv[nt]) * NW + ww) * (NTOK * HD);
      const float val = (acc[nt][i] + bv[nt]) * scl;
      // q,k row-major [tok][hd]; V transposed [hd][tok] (k-contig PV operand)
      const size_t off = (part == 2) ? (base + (size_t)hdiv[nt] * NTOK + ntok)
                                     : (base + (size_t)ntok * HD + hdiv[nt]);
      dst[off] = (_Float16)val;
    }
    if (++ntok == NTOK) { ntok = 0; if (++ww == NW) { ww = 0; ++b_; } }
  }
}

// ---------------------------------------------------------------------------
// Kernel 2: bias expansion. One block per (w, head); table column staged in
// LDS, then 20736 coalesced gathers -> f16 (15x batch reuse in attention).
// ---------------------------------------------------------------------------
__global__ __launch_bounds__(256) void ea3d_bias_kernel(
    const float* __restrict__ table, const int* __restrict__ pos_index,
    _Float16* __restrict__ biasbuf) {
  __shared__ float tcol[TABLE_ROWS];
  const int tid = threadIdx.x;
  const int w = blockIdx.x / HEADS, head = blockIdx.x % HEADS;
  for (int i = tid; i < TABLE_ROWS; i += 256)
    tcol[i] = table[(size_t)i * (NW * HEADS) + w * HEADS + head];
  __syncthreads();
  _Float16* dst = biasbuf + ((size_t)w * HEADS + head) * (NTOK * NTOK);
#pragma unroll
  for (int it = 0; it < (NTOK * NTOK) / 256; ++it) {   // 81 exact
    const int i = tid + it * 256;
    dst[i] = (_Float16)tcol[pos_index[i]];
  }
}

// ---------------------------------------------------------------------------
// Kernel 3: attention per (b, head, window). 9 waves; wave wv owns rows
// [16wv, 16wv+16). LDS overlay: {Q|K} tiles reused as prob buffer after sync.
// ---------------------------------------------------------------------------
__global__ __launch_bounds__(288) void ea3d_attn_kernel(
    const _Float16* __restrict__ qb, const _Float16* __restrict__ kb,
    const _Float16* __restrict__ vb, const float* __restrict__ mask,
    const _Float16* __restrict__ biasbuf, _Float16* __restrict__ ob) {
  __shared__ __align__(32) _Float16 smem[9 * 16 * NPAD]; // Q(4608)+K(4608) then P
  __shared__ __align__(32) _Float16 Vt[HD * NPAD];       // V^T [32][160], pad 0

  const int tid = threadIdx.x, lane = tid & 31, wv = tid >> 5;
  const int blk  = blockIdx.x;
  const int b    = blk / (HEADS * NW);
  const int rem  = blk % (HEADS * NW);
  const int head = rem / NW;
  const int w    = rem % NW;

  _Float16* Qs = smem;                 // row-major [tok][32]
  _Float16* Ks = smem + NTOK * HD;     // row-major [tok][32] (k-contig for B)
  const size_t qkvoff = (((size_t)b * HEADS + head) * NW + w) * (NTOK * HD);

  // ---- stage Q, K, V^T into LDS ----
  {
    const uint4* qs = (const uint4*)(qb + qkvoff);
    const uint4* ks = (const uint4*)(kb + qkvoff);
    uint4* Qw = (uint4*)Qs;
    uint4* Kw = (uint4*)Ks;
#if EA3D_ASYNC
#pragma unroll
    for (int it = 0; it < 2; ++it) {               // 576 b128 chunks each
      const int i = tid + it * 288;
      __builtin_amdgcn_global_load_async_to_lds_b128(gcast(qs + i),
                                                     lcast(Qw + i), 0, 0);
      __builtin_amdgcn_global_load_async_to_lds_b128(gcast(ks + i),
                                                     lcast(Kw + i), 0, 0);
    }
#pragma unroll
    for (int it = 0; it < 2; ++it) {               // V^T rows: 18 chunks x 32
      const int c = tid + it * 288;
      const int d = c / 18, j = c % 18;
      __builtin_amdgcn_global_load_async_to_lds_b128(
          gcast(vb + qkvoff + (size_t)d * NTOK + j * 8),
          lcast(&Vt[d * NPAD + j * 8]), 0, 0);
    }
    if (tid < 64) {                                // zero pad cols 144..159
      const int d = tid >> 1, j = tid & 1;
      uint4 zz; zz.x = zz.y = zz.z = zz.w = 0u;
      *(uint4*)&Vt[d * NPAD + NTOK + j * 8] = zz;
    }
    EA3D_WAIT_ASYNC();
#else
#pragma unroll
    for (int it = 0; it < 2; ++it) {
      const int i = tid + it * 288;
      Qw[i] = qs[i]; Kw[i] = ks[i];
    }
    const unsigned* vs = (const unsigned*)(vb + qkvoff);   // V^T [32][144]
    unsigned* Vw = (unsigned*)Vt;                          // V^T [32][160]
#pragma unroll
    for (int it = 0; it < 9; ++it) {               // 2560 words, last partial
      const int i = tid + it * 288;
      if (i < HD * NPAD / 2) {
        const int d = i / (NPAD / 2), wz = i % (NPAD / 2);
        Vw[i] = (wz < NTOK / 2) ? vs[d * (NTOK / 2) + wz] : 0u;
      }
    }
#endif
  }
  __syncthreads();

  const int hi = lane >> 4, nn = lane & 15;

  // ---- scores: S[16,144] = Q_strip @ K^T (9 WMMAs) ----
  v8f z = {};
  v8f acc[9];
#pragma unroll
  for (int t = 0; t < 9; ++t) acc[t] = z;
  {
    v16h aq = frag_a(&Qs[wv * 16 * HD], HD, lane);
#pragma unroll
    for (int t = 0; t < 9; ++t) {
      v16h bk = frag_b(&Ks[t * 16 * HD], HD, lane);  // B[k][n] = K[t*16+n][k]
      acc[t] = WMMA_F16(aq, bk, acc[t]);
    }
  }

  // ---- + precomputed earth-position bias + lon mask (both coalesced) ----
  const _Float16* bbase = biasbuf + ((size_t)w * HEADS + head) * (NTOK * NTOK);
  const float* mbase = mask + (((size_t)b * NW + w) * NTOK) * NTOK;
#pragma unroll
  for (int i = 0; i < 8; ++i) {
    const int q = wv * 16 + i + 8 * hi;
#pragma unroll
    for (int t = 0; t < 9; ++t) {
      const int kk = t * 16 + nn;
      acc[t][i] += (float)bbase[q * NTOK + kk] + mbase[q * NTOK + kk];
    }
  }

  // ---- softmax over k (144); row lives across the 16 lanes of a half ----
#pragma unroll
  for (int i = 0; i < 8; ++i) {
    float m = -3.0e38f;
#pragma unroll
    for (int t = 0; t < 9; ++t) m = fmaxf(m, acc[t][i]);
#pragma unroll
    for (int o = 1; o < 16; o <<= 1) m = fmaxf(m, __shfl_xor(m, o, 32));
    float s = 0.f;
#pragma unroll
    for (int t = 0; t < 9; ++t) { float e = __expf(acc[t][i] - m); acc[t][i] = e; s += e; }
#pragma unroll
    for (int o = 1; o < 16; o <<= 1) s += __shfl_xor(s, o, 32);
    const float inv = 1.0f / s;
#pragma unroll
    for (int t = 0; t < 9; ++t) acc[t][i] *= inv;
  }

  __syncthreads();   // all Q/K reads done -> safe to overlay prob buffer

  // ---- spill probs row-major [16][160] (A-fragment friendly) ----
  _Float16* Pw = smem + wv * 16 * NPAD;
#pragma unroll
  for (int i = 0; i < 8; ++i) {
    const int m = i + 8 * hi;
#pragma unroll
    for (int t = 0; t < 9; ++t) Pw[m * NPAD + t * 16 + nn] = (_Float16)acc[t][i];
  }
#pragma unroll
  for (int it = 0; it < 8; ++it) {               // zero pad cols 144..159
    const int j = lane + it * 32;
    Pw[(j >> 4) * NPAD + NTOK + (j & 15)] = (_Float16)0.f;
  }
  __syncthreads();

  // ---- PV: [16,160] @ [160,32] -> [16,32] (5 K-steps, 2 N-tiles) ----
  v8f o0 = z, o1 = z;
#pragma unroll
  for (int kt = 0; kt < 5; ++kt) {
    v16h ap  = frag_a(&Pw[kt * 32], NPAD, lane);
    v16h bv0 = frag_b(&Vt[0 * NPAD + kt * 32], NPAD, lane);   // cols 0..15
    v16h bv1 = frag_b(&Vt[16 * NPAD + kt * 32], NPAD, lane);  // cols 16..31
    o0 = WMMA_F16(ap, bv0, o0);
    o1 = WMMA_F16(ap, bv1, o1);
  }

  // ---- write [b,w,n, head*32 + d] so projection reads contiguous C ----
  const size_t obase = ((size_t)b * NW + w) * NTOK;
#pragma unroll
  for (int i = 0; i < 8; ++i) {
    const int q = wv * 16 + i + 8 * hi;
    const size_t o = (obase + q) * DIMC + head * HD;
    ob[o + nn]      = (_Float16)o0[i];
    ob[o + 16 + nn] = (_Float16)o1[i];
  }
}

// ---------------------------------------------------------------------------
// Kernel 4: output projection (f16 in / fp32 out + bias). Same async
// double-buffered GEMM as kernel 1.
// ---------------------------------------------------------------------------
__global__ __launch_bounds__(128) void ea3d_proj_kernel(
    const _Float16* __restrict__ in, const _Float16* __restrict__ wh,
    const float* __restrict__ bias, float* __restrict__ out) {
  __shared__ __align__(32) _Float16 As[2][64 * 32];   // row-major [m][k]
  __shared__ __align__(32) _Float16 Bs[2][64 * 32];   // n-major  [n][k]
  const int tid = threadIdx.x, lane = tid & 31, wv = tid >> 5;
  const int rowBase = blockIdx.x * 64;
  const int colBase = blockIdx.y * 64;

  auto stage = [&](int kt, int buf) {
#pragma unroll
    for (int it = 0; it < 2; ++it) {
      const int i = tid + it * 128;
      const int r = i >> 2, c8 = (i & 3) * 8;
#if EA3D_ASYNC
      __builtin_amdgcn_global_load_async_to_lds_b128(
          gcast(in + (size_t)(rowBase + r) * DIMC + kt * 32 + c8),
          lcast(&As[buf][r * 32 + c8]), 0, 0);
      __builtin_amdgcn_global_load_async_to_lds_b128(
          gcast(wh + (size_t)(colBase + r) * DIMC + kt * 32 + c8),
          lcast(&Bs[buf][r * 32 + c8]), 0, 0);
#else
      *(v8h*)&As[buf][r * 32 + c8] =
          *(const v8h*)(in + (size_t)(rowBase + r) * DIMC + kt * 32 + c8);
      *(v8h*)&Bs[buf][r * 32 + c8] =
          *(const v8h*)(wh + (size_t)(colBase + r) * DIMC + kt * 32 + c8);
#endif
    }
  };

  v8f z = {};
  v8f acc[4];
#pragma unroll
  for (int nt = 0; nt < 4; ++nt) acc[nt] = z;

  stage(0, 0);
  EA3D_WAIT_ASYNC();
  __syncthreads();
#pragma unroll
  for (int kt = 0; kt < 6; ++kt) {
    const int buf = kt & 1;
    if (kt < 5) stage(kt + 1, buf ^ 1);   // DMA overlaps WMMA below
    v16h a = frag_a(&As[buf][wv * 16 * 32], 32, lane);
#pragma unroll
    for (int nt = 0; nt < 4; ++nt) {
      v16h b = frag_b(&Bs[buf][nt * 16 * 32], 32, lane);
      acc[nt] = WMMA_F16(a, b, acc[nt]);
    }
    if (kt < 5) { EA3D_WAIT_ASYNC(); __syncthreads(); }
  }

  const int hi = lane >> 4, nn = lane & 15;
#pragma unroll
  for (int nt = 0; nt < 4; ++nt) {
    const int cg = colBase + nt * 16 + nn;
    const float bv = bias[cg];
#pragma unroll
    for (int i = 0; i < 8; ++i) {
      const int rg = rowBase + wv * 16 + i + 8 * hi;
      out[(size_t)rg * DIMC + cg] = acc[nt][i] + bv;
    }
  }
}

// ---------------------------------------------------------------------------
extern "C" void kernel_launch(void* const* d_in, const int* in_sizes, int n_in,
                              void* d_out, int out_size, void* d_ws, size_t ws_size,
                              hipStream_t stream) {
  (void)in_sizes; (void)n_in; (void)out_size; (void)ws_size;
  const float* x        = (const float*)d_in[0];
  const float* mask     = (const float*)d_in[1];
  const float* qkv_w    = (const float*)d_in[2];
  const float* qkv_b    = (const float*)d_in[3];
  const float* table    = (const float*)d_in[4];
  const float* proj_w   = (const float*)d_in[5];
  const float* proj_b   = (const float*)d_in[6];
  const int*   pos_idx  = (const int*)d_in[7];
  float* out = (float*)d_out;

  _Float16* qb  = (_Float16*)d_ws;
  _Float16* kb  = qb + QKV_ELEMS;
  _Float16* vb  = kb + QKV_ELEMS;
  _Float16* ob  = vb + QKV_ELEMS;
  _Float16* bbf = ob + QKV_ELEMS;
  _Float16* xh  = bbf + BIAS_ELEMS;                      // x as f16
  _Float16* wqh = xh + (size_t)MROWS * DIMC;             // qkv_w as f16
  _Float16* wph = wqh + (size_t)3 * DIMC * DIMC;         // proj_w as f16

  // K0: f32 -> f16 (exact grids: sizes are multiples of 2048)
  ea3d_cvt_kernel<<<dim3((MROWS * (size_t)DIMC) / 2048), 256, 0, stream>>>(x, xh);
  ea3d_cvt_kernel<<<dim3((3 * DIMC * DIMC) / 2048), 256, 0, stream>>>(qkv_w, wqh);
  ea3d_cvt_kernel<<<dim3((DIMC * DIMC) / 2048), 256, 0, stream>>>(proj_w, wph);

  ea3d_qkv_kernel<<<dim3(MROWS / 64, (3 * DIMC) / 64), 128, 0, stream>>>(
      xh, wqh, qkv_b, qb, kb, vb);
  ea3d_bias_kernel<<<dim3(NW * HEADS), 256, 0, stream>>>(table, pos_idx, bbf);
  ea3d_attn_kernel<<<dim3(BBATCH * HEADS * NW), 288, 0, stream>>>(
      qb, kb, vb, mask, bbf, ob);
  ea3d_proj_kernel<<<dim3(MROWS / 64, DIMC / 64), 128, 0, stream>>>(
      ob, proj_w ? wph : wph, proj_b, out);
}
